// RelativeNeuralNetworkAttentionMultiHead_69020124447483
// MI455X (gfx1250) — matrix operand order repaired and checked by reading, not compiled, not run on hardware
//
#include <hip/hip_runtime.h>
#include <hip/hip_bf16.h>

typedef __attribute__((ext_vector_type(16))) _Float16 v16h;
typedef __attribute__((ext_vector_type(8)))  _Float16 v8h;
typedef __attribute__((ext_vector_type(8)))  float    v8f;

// ---------------------------------------------------------------------------
// WMMA helpers (CDNA5 v_wmma_f32_16x16x32_f16, wave32)
// ---------------------------------------------------------------------------
__device__ __forceinline__ v8f wmma_f16(v16h a, v16h b, v8f c) {
  return __builtin_amdgcn_wmma_f32_16x16x32_f16(
      /*neg_a=*/false, a, /*neg_b=*/false, b,
      /*c_mod=*/(short)0, c, /*reuse_a=*/false, /*reuse_b=*/false);
}

__device__ __forceinline__ v16h frag_from2(const _Float16* p0, const _Float16* p1) {
  v16h r;
  v8h a = *(const v8h*)p0;
  v8h b = *(const v8h*)p1;
#pragma unroll
  for (int i = 0; i < 8; ++i) { r[i] = a[i]; r[i + 8] = b[i]; }
  return r;
}

// A fragment (16x32 f16) from row-major A[M][ld]; tile origin (m0,k0).
// Lane l: M = m0+(l&15); halves e=0..15 -> K = kb + e (contiguous),
// kb = k0 + (l<16 ? 0 : 16).
__device__ __forceinline__ v16h load_a_frag(const _Float16* A, int ld, int m0, int k0, int lane) {
  int m  = m0 + (lane & 15);
  int kb = k0 + ((lane < 16) ? 0 : 16);
  const _Float16* base = A + m * ld + kb;
  return frag_from2(base, base + 8);
}

// B fragment (32x16 f16) where BT[n][k] == B[k][n] (row-major, stride ld).
// Lane l: N = n0+(l&15); halves e=0..15 -> K = k0 + (l<16?0:16) + e  (contiguous).
__device__ __forceinline__ v16h load_b_fragT(const _Float16* BT, int ld, int k0, int n0, int lane) {
  int n  = n0 + (lane & 15);
  int kb = k0 + ((lane < 16) ? 0 : 16);
  const _Float16* base = BT + n * ld + kb;
  return frag_from2(base, base + 8);
}

// B fragment gathered from row-major B[k][n] (stride ld) when no transpose exists.
__device__ __forceinline__ v16h load_b_frag_gather(const _Float16* B, int ld, int k0, int n0, int lane) {
  int n  = n0 + (lane & 15);
  int kb = k0 + ((lane < 16) ? 0 : 16);
  v16h r;
#pragma unroll
  for (int e = 0; e < 16; ++e) r[e] = B[(kb + e) * ld + n];
  return r;
}

// ---------------------------------------------------------------------------
// Problem constants (b=8, t=128, m=127, k=128, h=4)
// ---------------------------------------------------------------------------
#define BT_TOTAL 1024   // b*t
#define MDIM 127
#define KDIM 128
#define HNUM 4

// workspace layout (bytes)
#define WS_GH   0            // _Float16 GH[4][128][256]  (G | H)   262144 B
#define WS_U    262144       // _Float16 u[4][128]                    1024 B
#define WS_V    263168       // _Float16 v[4][128]                    1024 B
#define WS_BVV  264192       // float    bvv[4][128]                  2048 B
#define WS_BV0  266240       // float    bv0[4]                         16 B

// ---------------------------------------------------------------------------
// Kernel 1: u[h,k] = sum_j Wk[j,k]*Wv[h*129, j]
//           v[h,k] = sum_j Wq[j,k]*Wv[h*129, 128+j]
//           bias rearrangement. Tiny; one 1024-thread block.
// ---------------------------------------------------------------------------
__global__ __launch_bounds__(1024) void precompute_uv(
    const float* __restrict__ Wk, const float* __restrict__ Wq,
    const float* __restrict__ Wv, const float* __restrict__ bv,
    _Float16* __restrict__ uF, _Float16* __restrict__ vF,
    float* __restrict__ bvv, float* __restrict__ bv0)
{
  const int tid = threadIdx.x;
  const int sel = tid >> 9;          // 0 -> u, 1 -> v
  const int h   = (tid >> 7) & 3;
  const int k   = tid & 127;
  const float* W   = sel ? Wq : Wk;
  const float* wvr = Wv + (size_t)(h * 129) * 256 + (sel ? 128 : 0);
  float s = 0.f;
#pragma unroll 8
  for (int j = 0; j < 128; ++j) s += W[j * 128 + k] * wvr[j];
  (sel ? vF : uF)[h * 128 + k] = (_Float16)s;

  if (tid < 512) bvv[tid] = bv[(tid >> 7) * 129 + 1 + (tid & 127)];
  if (tid < 4)   bv0[tid] = bv[tid * 129];
}

// ---------------------------------------------------------------------------
// Kernel 2: GH[r=h*128+kk][0:128]   = G = sum_c Wv[o, c]    * Wk[c, k]
//           GH[r=h*128+kk][128:256] = H = sum_j Wv[o,128+j] * Wq[j, k]
//           o = h*129+1+kk.  WMMA GEMM, 32 blocks x 16 rows.
// ---------------------------------------------------------------------------
__global__ __launch_bounds__(256) void precompute_GH(
    const float* __restrict__ Wk, const float* __restrict__ Wq,
    const float* __restrict__ Wv, _Float16* __restrict__ GH)
{
  __shared__ __attribute__((aligned(16))) _Float16 sA[16 * 256];    // Wv rows, f16
  __shared__ __attribute__((aligned(16))) _Float16 sBT[128 * 128];  // WkT / WqT, f16
  const int tid = threadIdx.x, lane = tid & 31, wave = tid >> 5;
  const int r0 = blockIdx.x * 16;   // output-row tile (512 rows total)

  // stage A tile (full 256-wide Wv rows for this row tile)
  for (int i = tid; i < 16 * 256; i += 256) {
    int rr = r0 + (i >> 8);
    int h = rr >> 7, kk = rr & 127;
    int o = h * 129 + 1 + kk;
    sA[i] = (_Float16)Wv[(size_t)o * 256 + (i & 255)];
  }
  // stage Wk^T: sBT[n][c] = Wk[c][n]
  for (int i = tid; i < 16384; i += 256) {
    int n = i >> 7, c = i & 127;
    sBT[i] = (_Float16)Wk[c * 128 + n];
  }
  __syncthreads();

  { // G part: A cols 0..127, output cols 0..127
    v8f acc = {};
    const int n0 = wave * 16;
#pragma unroll
    for (int kt = 0; kt < 4; ++kt) {
      v16h a = load_a_frag(sA, 256, 0, kt * 32, lane);
      v16h b = load_b_fragT(sBT, 128, kt * 32, n0, lane);
      acc = wmma_f16(a, b, acc);
    }
    const int n = n0 + (lane & 15);
    const int roff = (lane < 16) ? 0 : 8;
#pragma unroll
    for (int i = 0; i < 8; ++i)
      GH[(size_t)(r0 + roff + i) * 256 + n] = (_Float16)acc[i];
  }
  __syncthreads();
  // stage Wq^T
  for (int i = tid; i < 16384; i += 256) {
    int n = i >> 7, c = i & 127;
    sBT[i] = (_Float16)Wq[c * 128 + n];
  }
  __syncthreads();
  { // H part: A cols 128..255, output cols 128..255
    v8f acc = {};
    const int n0 = wave * 16;
#pragma unroll
    for (int kt = 0; kt < 4; ++kt) {
      v16h a = load_a_frag(sA, 256, 0, 128 + kt * 32, lane);
      v16h b = load_b_fragT(sBT, 128, kt * 32, n0, lane);
      acc = wmma_f16(a, b, acc);
    }
    const int n = n0 + (lane & 15);
    const int roff = (lane < 16) ? 0 : 8;
#pragma unroll
    for (int i = 0; i < 8; ++i)
      GH[(size_t)(r0 + roff + i) * 256 + 128 + n] = (_Float16)acc[i];
  }
}

// ---------------------------------------------------------------------------
// Kernel 3: one block per (b,t). HBM-bound: streams the 127x128 kv tile once.
//   score[m,h] = kv[m]·u[h] + q·v[h] + bv0[h]          (WMMA, N padded to 16)
//   w = softmax_m(score)                               (wave-parallel, v_exp)
//   p[h,k] = sum_m w[m,h]*kv[m,k]                      (WMMA, M padded to 16)
//   out[h,kk] = bvv + [p[h] | q] · GH[h][kk][:]        (WMMA, K=256)
// ---------------------------------------------------------------------------
__global__ __launch_bounds__(256) void attn_main(
    const float* __restrict__ q_x, const float* __restrict__ kv_x,
    const _Float16* __restrict__ GH, const _Float16* __restrict__ uF,
    const _Float16* __restrict__ vF, const float* __restrict__ bvv,
    const float* __restrict__ bv0, float* __restrict__ out)
{
  __shared__ __attribute__((aligned(16))) _Float16 sKV[128 * 128];  // kv tile f16 (row 127 = 0)
  __shared__ __attribute__((aligned(16))) _Float16 sUB[16 * 128];   // u rows 0-3, zeros 4-15
  __shared__ __attribute__((aligned(16))) _Float16 sWT[16 * 128];   // softmax weights^T, padded
  __shared__ __attribute__((aligned(16))) _Float16 sPQA[16 * 256];  // rows 0-3: [p[h] | q]
  __shared__ __attribute__((aligned(16))) _Float16 sQ[128];
  __shared__ float sQf[128];
  __shared__ float sScore[4 * 128];
  __shared__ float sQs[4];

  const int tid  = threadIdx.x;
  const int lane = tid & 31;
  const int wave = tid >> 5;
  const int bt   = blockIdx.x;                         // b*128 + t
  const size_t kv_base = (size_t)bt * MDIM * KDIM;

  // ---- Phase 1a: stage inputs ----
  if (tid < 128) {
    float qv = q_x[(size_t)bt * 128 + tid];
    sQf[tid] = qv;
    sQ[tid]  = (_Float16)qv;
  }
  for (int idx = tid; idx < MDIM * 32; idx += 256) {   // 4064 float4 loads
    int m  = idx >> 5;
    int c4 = (idx & 31) << 2;
    const float4 f = *(const float4*)(kv_x + kv_base + (size_t)m * 128 + c4);
    _Float16* dst = sKV + m * 128 + c4;
    dst[0] = (_Float16)f.x; dst[1] = (_Float16)f.y;
    dst[2] = (_Float16)f.z; dst[3] = (_Float16)f.w;
  }
  for (int j = tid; j < 128; j += 256) sKV[127 * 128 + j] = (_Float16)0.f;
  for (int i = tid; i < 2048; i += 256) {
    int r = i >> 7, c = i & 127;
    sUB[i] = (r < 4) ? uF[r * 128 + c] : (_Float16)0.f;
    sWT[i] = (_Float16)0.f;
  }
  for (int i = tid; i < 4096; i += 256) sPQA[i] = (_Float16)0.f;
  __syncthreads();

  // ---- Phase 1b: per-head query score + q half of the final A matrix ----
  if (tid < 4) {
    float s = 0.f;
    const _Float16* vr = vF + tid * 128;
#pragma unroll 8
    for (int k2 = 0; k2 < 128; ++k2) s += sQf[k2] * (float)vr[k2];
    sQs[tid] = s + bv0[tid];
  }
  for (int i = tid; i < 512; i += 256) {
    int r = i >> 7, c = i & 127;
    sPQA[r * 256 + 128 + c] = sQ[c];
  }
  __syncthreads();

  // ---- Phase 2: scores (128x128 kv) @ (128x16 u^T), one M-tile per wave ----
  {
    v8f acc = {};
    const int m0 = wave * 16;
#pragma unroll
    for (int kt = 0; kt < 4; ++kt) {
      v16h a = load_a_frag(sKV, 128, m0, kt * 32, lane);
      v16h b = load_b_fragT(sUB, 128, kt * 32, 0, lane);
      acc = wmma_f16(a, b, acc);
    }
    const int h = lane & 15;
    const int mrow = m0 + ((lane < 16) ? 0 : 8);
    if (h < 4) {
#pragma unroll
      for (int i = 0; i < 8; ++i) sScore[h * 128 + mrow + i] = acc[i] + sQs[h];
    }
  }
  __syncthreads();

  // ---- Phase 3: softmax over m (127 valid), one wave per head ----
  if (wave < 4) {
    const int h = wave;
    float vals[4];
    float mx = -3.4e38f;
#pragma unroll
    for (int it = 0; it < 4; ++it) {
      int m = lane + it * 32;
      float s = (m < MDIM) ? sScore[h * 128 + m] : -3.4e38f;
      vals[it] = s;
      mx = fmaxf(mx, s);
    }
#pragma unroll
    for (int d = 16; d > 0; d >>= 1) mx = fmaxf(mx, __shfl_xor(mx, d, 32));
    float sum = 0.f;
#pragma unroll
    for (int it = 0; it < 4; ++it) {
      int m = lane + it * 32;
      float e = (m < MDIM) ? __expf(vals[it] - mx) : 0.f;
      vals[it] = e;
      sum += e;
    }
#pragma unroll
    for (int d = 16; d > 0; d >>= 1) sum += __shfl_xor(sum, d, 32);
    const float inv = 1.f / sum;
#pragma unroll
    for (int it = 0; it < 4; ++it) {
      int m = lane + it * 32;
      if (m < MDIM) sWT[h * 128 + m] = (_Float16)(vals[it] * inv);
    }
  }
  __syncthreads();

  // ---- Phase 4: p = w^T (16x128m) @ kv (128m x 128k), one N-tile per wave ----
  {
    v8f acc = {};
    const int n0 = wave * 16;
#pragma unroll
    for (int kt = 0; kt < 4; ++kt) {
      v16h a = load_a_frag(sWT, 128, 0, kt * 32, lane);
      v16h b = load_b_frag_gather(sKV, 128, kt * 32, n0, lane);
      acc = wmma_f16(a, b, acc);
    }
    if (lane < 16) {
      const int kcol = n0 + lane;
#pragma unroll
      for (int h = 0; h < HNUM; ++h) sPQA[h * 256 + kcol] = (_Float16)acc[h];
    }
  }
  __syncthreads();

  // ---- Phase 5: out[h, n] = bvv + [p[h]|q] (K=256) · GH[h][n][:] ----
  {
    v16h afr[8];
#pragma unroll
    for (int kt = 0; kt < 8; ++kt) afr[kt] = load_a_frag(sPQA, 256, 0, kt * 32, lane);
    const int n0 = wave * 16;
    float* outbt = out + (size_t)bt * 512;
#pragma unroll
    for (int h = 0; h < HNUM; ++h) {
      v8f acc = {};
      const _Float16* GHh = GH + (size_t)h * 128 * 256;
#pragma unroll
      for (int kt = 0; kt < 8; ++kt) {
        v16h b = load_b_fragT(GHh, 256, kt * 32, n0, lane);
        acc = wmma_f16(afr[kt], b, acc);
      }
      if (lane < 16) {
        const int n = n0 + lane;
        outbt[h * 128 + n] = acc[h] + bvv[h * 128 + n];
      }
    }
  }
}

// ---------------------------------------------------------------------------
extern "C" void kernel_launch(void* const* d_in, const int* in_sizes, int n_in,
                              void* d_out, int out_size, void* d_ws, size_t ws_size,
                              hipStream_t stream) {
  (void)in_sizes; (void)n_in; (void)out_size; (void)ws_size;
  const float* q_x  = (const float*)d_in[0];
  const float* kv_x = (const float*)d_in[1];
  const float* Wk   = (const float*)d_in[2];
  const float* Wq   = (const float*)d_in[3];
  const float* Wv   = (const float*)d_in[4];
  const float* bv   = (const float*)d_in[5];
  float* out = (float*)d_out;

  char* ws = (char*)d_ws;
  _Float16* GH  = (_Float16*)(ws + WS_GH);
  _Float16* uF  = (_Float16*)(ws + WS_U);
  _Float16* vF  = (_Float16*)(ws + WS_V);
  float*    bvv = (float*)(ws + WS_BVV);
  float*    bv0 = (float*)(ws + WS_BV0);

  precompute_uv<<<1, 1024, 0, stream>>>(Wk, Wq, Wv, bv, uF, vF, bvv, bv0);
  precompute_GH<<<32, 256, 0, stream>>>(Wk, Wq, Wv, GH);
  attn_main<<<BT_TOTAL, 256, 0, stream>>>(q_x, kv_x, GH, uF, vF, bvv, bv0, out);
}
